// rnn_decoder_31756988187316
// MI455X (gfx1250) — matrix operand order repaired and checked by reading, not compiled, not run on hardware
//
#include <hip/hip_runtime.h>
#include <stdint.h>

// ---------------- problem dims ----------------
static constexpr int  Vn = 50000, En = 512, Hn = 512, Ln = 2, Bn = 32, Tn = 64, Sn = 100;
static constexpr int  Mtot = Bn * Tn;               // 2048 rows of the big GEMM
static constexpr long OUT_H   = (long)Bn * Tn * Vn;            // 102,400,000
static constexpr long OUT_C   = OUT_H + (long)Ln * Bn * Hn;    // +32768
static constexpr long OUT_ATT = OUT_C + (long)Ln * Bn * Hn;    // +32768

// ---------------- WMMA types ----------------
typedef __attribute__((ext_vector_type(16))) __bf16 v16bf;
typedef __attribute__((ext_vector_type(8)))  float  v8f;

union BFrag { v16bf v; uint4 q[2]; unsigned u[8]; };

__device__ __forceinline__ v8f wmma_bb(v16bf a, v16bf b, v8f c) {
  return __builtin_amdgcn_wmma_f32_16x16x32_bf16(false, a, false, b, (short)0, c, false, false);
}

// round-to-nearest-even f32 -> bf16
__device__ __forceinline__ unsigned short to_bf16(float x) {
  unsigned a = __builtin_bit_cast(unsigned, x);
  a = (a + 0x7FFFu + ((a >> 16) & 1u)) >> 16;
  return (unsigned short)a;
}
__device__ __forceinline__ unsigned pk2(float lo, float hi) {
  unsigned a = __builtin_bit_cast(unsigned, lo);
  unsigned b = __builtin_bit_cast(unsigned, hi);
  a = (a + 0x7FFFu + ((a >> 16) & 1u)) >> 16;
  b = (b + 0x7FFFu + ((b >> 16) & 1u)) >> 16;
  return (b << 16) | (a & 0xFFFFu);
}

// A fragment (16x32 MxK, bf16): lane<16 -> K=kb..kb+15,
// lane>=16 -> K=kb+16..kb+31 (contiguous); row = base + (lane&15).
__device__ __forceinline__ v16bf load_a_frag(const unsigned short* base, int ld, int row,
                                             int kb, int lane) {
  BFrag f;
  const int h8 = (lane & 16) ? 16 : 0;
  const unsigned short* p = base + (long)row * ld + kb + h8;
  f.q[0] = *(const uint4*)p;
  f.q[1] = *(const uint4*)(p + 8);
  return f.v;
}

// B fragment (32x16 KxN, bf16, weight stored row-major [N][K]):
// lane%16 = n, lane<16 -> K=kb..kb+15 ; lane>=16 -> K=kb+16..kb+31 (contiguous).
__device__ __forceinline__ v16bf load_b_frag(const unsigned short* base, int ld, long row,
                                             int kb, int lane) {
  BFrag f;
  const int h16 = (lane & 16) ? 16 : 0;
  const unsigned short* p = base + row * (long)ld + kb + h16;
  f.q[0] = *(const uint4*)p;
  f.q[1] = *(const uint4*)(p + 8);
  return f.v;
}

// Same B fragment but converting from f32 weights on the fly (fallback when the
// workspace cannot hold a pre-converted bf16 W_lin).
__device__ __forceinline__ v16bf load_b_frag_f32(const float* base, int ld, long row,
                                                 int kb, int lane) {
  BFrag f;
  const int h16 = (lane & 16) ? 16 : 0;
  const float4* p = (const float4*)(base + row * (long)ld + kb + h16);
  float4 x0 = p[0], x1 = p[1], x2 = p[2], x3 = p[3];
  f.u[0] = pk2(x0.x, x0.y); f.u[1] = pk2(x0.z, x0.w);
  f.u[2] = pk2(x1.x, x1.y); f.u[3] = pk2(x1.z, x1.w);
  f.u[4] = pk2(x2.x, x2.y); f.u[5] = pk2(x2.z, x2.w);
  f.u[6] = pk2(x3.x, x3.y); f.u[7] = pk2(x3.z, x3.w);
  return f.v;
}

__device__ __forceinline__ float sigm(float x)   { return 1.f / (1.f + __expf(-x)); }
__device__ __forceinline__ float tanh_f(float x) {
  float e = __expf(-2.f * fabsf(x));
  float r = (1.f - e) / (1.f + e);
  return copysignf(r, x);
}

// ---------------- phase 0: f32 -> bf16 weight conversion ----------------
__global__ void cvt_bf16_kernel(const float* __restrict__ src,
                                unsigned short* __restrict__ dst, int n) {
  for (int i = blockIdx.x * blockDim.x + threadIdx.x; i < n; i += gridDim.x * blockDim.x)
    dst[i] = to_bf16(src[i]);
}

// ---------------- phase 1: sequential LSTM + attention (one workgroup) ----------------
// 512 threads = 16 waves (4 waves/SIMD -> 256-VGPR budget).  Wave w processes
// hidden slices u = uh*256 + 16w .. +15 for uh = 0,1 SEQUENTIALLY so only
// acc[4][2] (64 VGPRs) is live per pass.  Cell state c lives in the workspace
// (L2-resident f32).  K-loops capped at unroll 2 (round-4 full-unroll spilled);
// the 4 gate B-fragments are loaded into distinct locals so one L2 round-trip
// covers 8 WMMAs instead of 2 (round-5 asm reused one B register pair).
// LDS: abf  [32][1024] bf16  @0       (x for layer0 in cols 0..511; attention cat rows)
//      hbuf [2][2][32][512] bf16 @64K (ping-pong [buf][layer] recurrent h)
//      gbuf [32][512] f32   @192K    (gamma)
//      sc   [32][112] f32   @256K    (scores / softmax weights)     total 270KB of 320KB
__global__ __launch_bounds__(512) void phase1_kernel(
    const int*   __restrict__ inputs, const float* __restrict__ h0,
    const float* __restrict__ c0,     const float* __restrict__ ctx,
    const float* __restrict__ emb,
    const float* __restrict__ bih,    const float* __restrict__ bhh,
    const float* __restrict__ bin,    const float* __restrict__ bout,
    const unsigned short* __restrict__ WihB, const unsigned short* __restrict__ WhhB,
    const unsigned short* __restrict__ WinB, const unsigned short* __restrict__ WoutB,
    float* __restrict__ cws,
    unsigned short* __restrict__ attbf, float* __restrict__ dout) {
  extern __shared__ char smem[];
  unsigned short* abf  = (unsigned short*)smem;               // 65536 B
  unsigned short* hbuf = (unsigned short*)(smem + 65536);     // 131072 B: [buf][layer][32][512]
  float*          gbuf = (float*)(smem + 196608);             // 65536 B
  float*          sc   = (float*)(smem + 262144);             // 14336 B

  const int tid  = threadIdx.x;
  const int lane = tid & 31;
  const int w    = tid >> 5;            // wave id 0..15
  const int lm   = lane & 15;
  const int hi8  = (lane & 16) ? 8 : 0; // C/D row offset for upper lane half

  // ---- init: h0 -> hbuf[0] (bf16), c0 -> workspace (f32, owned slots, no races)
  for (int idx = tid; idx < Ln * Bn * Hn; idx += 512) {
    hbuf[idx] = to_bf16(h0[idx]);
    cws[idx]  = c0[idx];
  }
  __syncthreads();

  for (int t = 0; t < Tn; ++t) {
    const int p = t & 1;
    const unsigned short* hcur = hbuf + p * (Ln * Bn * Hn);
    unsigned short*       hnew = hbuf + (p ^ 1) * (Ln * Bn * Hn);

    // ---- embedding gather -> abf[:, 0:512] (bf16)
    for (int idx = tid; idx < Bn * En; idx += 512) {
      const int b = idx >> 9, k = idx & 511;
      const int tok = inputs[b * Tn + t];
      abf[b * 1024 + k] = to_bf16(emb[(long)tok * En + k]);
    }
    __syncthreads();

    // ---- two LSTM layers
    #pragma unroll
    for (int l = 0; l < Ln; ++l) {
      const unsigned short* Ax  = (l == 0) ? abf : (hnew + 0 * Bn * Hn); // layer1 x = new h of layer0
      const int             ldx = (l == 0) ? 1024 : 512;
      const unsigned short* Ah  = hcur + l * Bn * Hn;
      #pragma unroll
      for (int uh = 0; uh < 2; ++uh) {
        const int u = uh * 256 + w * 16 + lm;
        v8f acc[4][2] = {};   // [gate][mt]
        #pragma unroll
        for (int src = 0; src < 2; ++src) {
          const unsigned short* Ab = src ? Ah : Ax;
          const int lda            = src ? 512 : ldx;
          const unsigned short* Bb = (src ? WhhB : WihB) + (long)l * 4 * Hn * Hn;
          #pragma unroll 2
          for (int kb = 0; kb < 512; kb += 32) {
            v16bf a0 = load_a_frag(Ab, lda, lm, kb, lane);
            v16bf a1 = load_a_frag(Ab, lda, 16 + lm, kb, lane);
            // all 4 gate B-fragments in flight together: 1 wait per 8 WMMAs
            v16bf q0 = load_b_frag(Bb, 512, 0 * 512 + u, kb, lane);
            v16bf q1 = load_b_frag(Bb, 512, 1 * 512 + u, kb, lane);
            v16bf q2 = load_b_frag(Bb, 512, 2 * 512 + u, kb, lane);
            v16bf q3 = load_b_frag(Bb, 512, 3 * 512 + u, kb, lane);
            acc[0][0] = wmma_bb(a0, q0, acc[0][0]);
            acc[0][1] = wmma_bb(a1, q0, acc[0][1]);
            acc[1][0] = wmma_bb(a0, q1, acc[1][0]);
            acc[1][1] = wmma_bb(a1, q1, acc[1][1]);
            acc[2][0] = wmma_bb(a0, q2, acc[2][0]);
            acc[2][1] = wmma_bb(a1, q2, acc[2][1]);
            acc[3][0] = wmma_bb(a0, q3, acc[3][0]);
            acc[3][1] = wmma_bb(a1, q3, acc[3][1]);
          }
        }
        // pointwise LSTM for this slice; writes only hnew / abf[:,512:] / cws which
        // no in-flight GEMM of this layer reads -> no barrier needed here.
        const float b0 = bih[l * 2048 + 0 * 512 + u] + bhh[l * 2048 + 0 * 512 + u];
        const float b1 = bih[l * 2048 + 1 * 512 + u] + bhh[l * 2048 + 1 * 512 + u];
        const float b2 = bih[l * 2048 + 2 * 512 + u] + bhh[l * 2048 + 2 * 512 + u];
        const float b3 = bih[l * 2048 + 3 * 512 + u] + bhh[l * 2048 + 3 * 512 + u];
        #pragma unroll
        for (int mt = 0; mt < 2; ++mt)
          #pragma unroll
          for (int e = 0; e < 8; ++e) {
            const int m = mt * 16 + hi8 + e;
            const long ci = (long)(l * Bn + m) * Hn + u;
            const float iv = sigm(acc[0][mt][e] + b0);
            const float fv = sigm(acc[1][mt][e] + b1);
            const float gv = tanh_f(acc[2][mt][e] + b2);
            const float ov = sigm(acc[3][mt][e] + b3);
            const float cn = fv * cws[ci] + iv * gv;
            cws[ci] = cn;
            const float hn = tanh_f(cn) * ov;
            const unsigned short hb = to_bf16(hn);
            hnew[l * (Bn * Hn) + m * 512 + u] = hb;
            if (l == Ln - 1) abf[m * 1024 + 512 + u] = hb;   // "out" half of attention cat
            if (t == Tn - 1) {
              dout[OUT_H + ci] = hn;
              dout[OUT_C + ci] = cn;
            }
          }
      }
      __syncthreads();   // hnew[l] complete before next layer / attention reads it
    }

    // ---- gamma = out @ W_in.T + b_in   (out = hnew[1])
    {
      const unsigned short* hsrc = hnew + 1 * Bn * Hn;
      #pragma unroll
      for (int uh = 0; uh < 2; ++uh) {
        const int u = uh * 256 + w * 16 + lm;
        v8f ga[2] = {};
        #pragma unroll 2
        for (int kb = 0; kb < 512; kb += 32) {
          v16bf a0 = load_a_frag(hsrc, 512, lm, kb, lane);
          v16bf a1 = load_a_frag(hsrc, 512, 16 + lm, kb, lane);
          v16bf bq = load_b_frag(WinB, 512, u, kb, lane);
          ga[0] = wmma_bb(a0, bq, ga[0]);
          ga[1] = wmma_bb(a1, bq, ga[1]);
        }
        const float bb = bin[u];
        #pragma unroll
        for (int mt = 0; mt < 2; ++mt)
          #pragma unroll
          for (int e = 0; e < 8; ++e) {
            const int m = mt * 16 + hi8 + e;
            gbuf[m * 512 + u] = ga[mt][e] + bb;
          }
      }
    }
    __syncthreads();

    // ---- scores[b][s] = contexts[b,s,:] . gamma[b,:]
    for (int idx = tid; idx < Bn * Sn; idx += 512) {
      const int b = idx / Sn, s = idx - b * Sn;
      const float4* cp = (const float4*)(ctx + (long)(b * Sn + s) * Hn);
      const float4* gp = (const float4*)(gbuf + b * Hn);
      float a = 0.f;
      #pragma unroll 4
      for (int k = 0; k < Hn / 4; ++k) {
        float4 c4 = cp[k], g4 = gp[k];
        a += c4.x * g4.x + c4.y * g4.y + c4.z * g4.z + c4.w * g4.w;
      }
      sc[b * 112 + s] = a;
    }
    __syncthreads();

    // ---- softmax over S (one lane per batch row), emit attns
    if (tid < Bn) {
      const int b = tid;
      float mx = -1e30f;
      for (int s = 0; s < Sn; ++s) mx = fmaxf(mx, sc[b * 112 + s]);
      float sum = 0.f;
      for (int s = 0; s < Sn; ++s) { float e = __expf(sc[b * 112 + s] - mx); sc[b * 112 + s] = e; sum += e; }
      const float inv = 1.f / sum;
      for (int s = 0; s < Sn; ++s) {
        const float wv = sc[b * 112 + s] * inv;
        sc[b * 112 + s] = wv;
        dout[OUT_ATT + (long)(b * Tn + t) * Sn + s] = wv;
      }
    }
    __syncthreads();

    // ---- c_t = w @ contexts -> abf[:, 0:512] (bf16 half of cat)
    for (int idx = tid; idx < Bn * Hn; idx += 512) {
      const int b = idx >> 9, uu = idx & 511;
      const float* cb = ctx + (long)b * Sn * Hn + uu;
      const float* wp = sc + b * 112;
      float a = 0.f;
      for (int s = 0; s < Sn; ++s) a += wp[s] * cb[(long)s * Hn];
      abf[b * 1024 + uu] = to_bf16(a);
    }
    __syncthreads();

    // ---- att_out = tanh(cat @ W_out.T + b_out) -> ws row (b*T + t), bf16
    {
      #pragma unroll
      for (int uh = 0; uh < 2; ++uh) {
        const int u = uh * 256 + w * 16 + lm;
        v8f ao[2] = {};
        #pragma unroll 2
        for (int kb = 0; kb < 1024; kb += 32) {
          v16bf a0 = load_a_frag(abf, 1024, lm, kb, lane);
          v16bf a1 = load_a_frag(abf, 1024, 16 + lm, kb, lane);
          v16bf bq = load_b_frag(WoutB, 1024, u, kb, lane);
          ao[0] = wmma_bb(a0, bq, ao[0]);
          ao[1] = wmma_bb(a1, bq, ao[1]);
        }
        const float bb = bout[u];
        #pragma unroll
        for (int mt = 0; mt < 2; ++mt)
          #pragma unroll
          for (int e = 0; e < 8; ++e) {
            const int m = mt * 16 + hi8 + e;   // == batch index b
            const float x = tanh_f(ao[mt][e] + bb);
            attbf[(long)(m * Tn + t) * Hn + u] = to_bf16(x);
          }
      }
    }
    __syncthreads();   // before next step's embedding overwrites abf
  }
}

// ---------------- phase 2: logits = att_out @ W_lin.T + b_lin ----------------
// Grid: x = N blocks of 256 (196), y = M blocks of 64 (32). 8 waves as 2(M)x4(N);
// each wave computes 32x64 via 8 WMMA accumulators. W_lin stays L2-resident
// (102MB f32 / 51MB bf16 vs 192MB L2). PRE=true uses a pre-converted bf16 copy.
// N-edge handling: tile rows are clamped (V multiple of 16) so the K-loop is
// branch-free; out-of-range tiles compute garbage that is never stored.
template <bool PRE>
__global__ __launch_bounds__(256) void phase2_kernel(
    const unsigned short* __restrict__ attbf, const unsigned short* __restrict__ WlinB,
    const float* __restrict__ Wlin, const float* __restrict__ blin,
    float* __restrict__ out) {
  const int tid  = threadIdx.x;
  const int lane = tid & 31;
  const int wv   = tid >> 5;
  const int wM   = wv >> 2, wN = wv & 3;
  const int  Mb  = blockIdx.y * 64 + wM * 32;
  const long Nb  = (long)blockIdx.x * 256 + wN * 64;
  const int lm   = lane & 15;
  const int hi8  = (lane & 16) ? 8 : 0;

  // clamped tile base rows (keep loads in-bounds; stores are guarded below)
  long r0 = Nb +  0; if (r0 >= Vn) r0 = Vn - 16;
  long r1 = Nb + 16; if (r1 >= Vn) r1 = Vn - 16;
  long r2 = Nb + 32; if (r2 >= Vn) r2 = Vn - 16;
  long r3 = Nb + 48; if (r3 >= Vn) r3 = Vn - 16;

  v8f acc[2][4] = {};
  #pragma unroll 2
  for (int kb = 0; kb < 512; kb += 32) {
    v16bf a0 = load_a_frag(attbf, Hn, Mb + lm, kb, lane);
    v16bf a1 = load_a_frag(attbf, Hn, Mb + 16 + lm, kb, lane);
    v16bf q0, q1, q2, q3;
    if constexpr (PRE) {
      q0 = load_b_frag(WlinB, Hn, r0 + lm, kb, lane);
      q1 = load_b_frag(WlinB, Hn, r1 + lm, kb, lane);
      q2 = load_b_frag(WlinB, Hn, r2 + lm, kb, lane);
      q3 = load_b_frag(WlinB, Hn, r3 + lm, kb, lane);
    } else {
      q0 = load_b_frag_f32(Wlin, Hn, r0 + lm, kb, lane);
      q1 = load_b_frag_f32(Wlin, Hn, r1 + lm, kb, lane);
      q2 = load_b_frag_f32(Wlin, Hn, r2 + lm, kb, lane);
      q3 = load_b_frag_f32(Wlin, Hn, r3 + lm, kb, lane);
    }
    acc[0][0] = wmma_bb(a0, q0, acc[0][0]);
    acc[1][0] = wmma_bb(a1, q0, acc[1][0]);
    acc[0][1] = wmma_bb(a0, q1, acc[0][1]);
    acc[1][1] = wmma_bb(a1, q1, acc[1][1]);
    acc[0][2] = wmma_bb(a0, q2, acc[0][2]);
    acc[1][2] = wmma_bb(a1, q2, acc[1][2]);
    acc[0][3] = wmma_bb(a0, q3, acc[0][3]);
    acc[1][3] = wmma_bb(a1, q3, acc[1][3]);
  }
  #pragma unroll
  for (int nt = 0; nt < 4; ++nt) {
    const long n0 = Nb + nt * 16;
    if (n0 >= Vn) continue;
    const long n = n0 + lm;
    const float bias = blin[n];
    #pragma unroll
    for (int mt = 0; mt < 2; ++mt)
      #pragma unroll
      for (int e = 0; e < 8; ++e) {
        const long m = Mb + mt * 16 + hi8 + e;
        out[m * Vn + n] = acc[mt][nt][e] + bias;
      }
  }
}

// ---------------- host ----------------
extern "C" void kernel_launch(void* const* d_in, const int* in_sizes, int n_in,
                              void* d_out, int out_size, void* d_ws, size_t ws_size,
                              hipStream_t stream) {
  const int*   inputs = (const int*)d_in[0];
  const float* h0     = (const float*)d_in[1];
  const float* c0     = (const float*)d_in[2];
  const float* ctx    = (const float*)d_in[3];
  const float* emb    = (const float*)d_in[4];
  const float* Wih    = (const float*)d_in[5];
  const float* Whh    = (const float*)d_in[6];
  const float* bih    = (const float*)d_in[7];
  const float* bhh    = (const float*)d_in[8];
  const float* Win    = (const float*)d_in[9];
  const float* bin    = (const float*)d_in[10];
  const float* Wout   = (const float*)d_in[11];
  const float* bout   = (const float*)d_in[12];
  const float* Wlin   = (const float*)d_in[13];
  const float* blin   = (const float*)d_in[14];
  float* dout = (float*)d_out;

  // workspace layout (bytes)
  char* ws = (char*)d_ws;
  unsigned short* attbf = (unsigned short*)(ws);               // 2048*512*2   = 2,097,152
  unsigned short* WihB  = (unsigned short*)(ws + 2097152);     // 2*2048*512*2 = 4,194,304
  unsigned short* WhhB  = (unsigned short*)(ws + 6291456);     // 4,194,304
  unsigned short* WinB  = (unsigned short*)(ws + 10485760);    // 512*512*2    = 524,288
  unsigned short* WoutB = (unsigned short*)(ws + 11010048);    // 512*1024*2   = 1,048,576
  float*          cws   = (float*)(ws + 12058624);             // 2*32*512*4   = 131,072
  unsigned short* WlinB = (unsigned short*)(ws + 12189696);    // 50000*512*2  = 51,200,000
  const bool pre = ws_size >= (size_t)12189696 + 51200000;     // fixed per run -> deterministic

  // phase 0: weight conversions (parallel, cheap: ~160MB total HBM traffic ~ 7us)
  cvt_bf16_kernel<<<1024, 256, 0, stream>>>(Wih,  WihB,  Ln * 4 * Hn * En);
  cvt_bf16_kernel<<<1024, 256, 0, stream>>>(Whh,  WhhB,  Ln * 4 * Hn * Hn);
  cvt_bf16_kernel<<<256,  256, 0, stream>>>(Win,  WinB,  Hn * Hn);
  cvt_bf16_kernel<<<512,  256, 0, stream>>>(Wout, WoutB, Hn * 2 * Hn);
  if (pre)
    cvt_bf16_kernel<<<4096, 256, 0, stream>>>(Wlin, WlinB, Vn * Hn);

  // phase 1: sequential recurrence (single WG, 16 waves, ~270KB LDS of the 320KB WGP pool)
  const size_t lds = 65536 + 131072 + 65536 + 14336;
  phase1_kernel<<<1, 512, lds, stream>>>(inputs, h0, c0, ctx, emb, bih, bhh, bin, bout,
                                         WihB, WhhB, WinB, WoutB, cws, attbf, dout);

  // phase 2: big vocab GEMM (105 GFLOP, 6272 WGs)
  dim3 grid((Vn + 255) / 256, Mtot / 64, 1);
  if (pre)
    phase2_kernel<true><<<grid, 256, 0, stream>>>(attbf, WlinB, Wlin, blin, dout);
  else
    phase2_kernel<false><<<grid, 256, 0, stream>>>(attbf, WlinB, Wlin, blin, dout);
}
